// VQVAE_23476291240378
// MI455X (gfx1250) — matrix operand order repaired and checked by reading, not compiled, not run on hardware
//
#include <hip/hip_runtime.h>
#include <hip/hip_bf16.h>
#include <math.h>

// ---------------- WMMA vector types (gfx1250 / CDNA5) ----------------
typedef __attribute__((ext_vector_type(16))) __bf16 v16bf;
typedef __attribute__((ext_vector_type(8)))  __bf16 v8bf;
typedef __attribute__((ext_vector_type(8)))  float  v8f;

// RNE f32 -> bf16 via bit trick (finite inputs only, which holds here)
__device__ __forceinline__ __bf16 f2bf(float f) {
  unsigned u = __builtin_bit_cast(unsigned, f);
  u += 0x7FFFu + ((u >> 16) & 1u);
  unsigned short h = (unsigned short)(u >> 16);
  return __builtin_bit_cast(__bf16, h);
}

// ---------------- f32 -> bf16 weight conversion ----------------
__global__ __launch_bounds__(256) void cvt_bf16_kernel(const float* __restrict__ src,
                                                       __bf16* __restrict__ dst, int n) {
  int i = blockIdx.x * 256 + threadIdx.x;
  if (i < n) dst[i] = f2bf(src[i]);
}

// A-fragment (16x32 bf16): lanes 0-15 hold K[kb..kb+7] then K[kb+8..kb+15]
// (kb is pre-biased +16 for lanes 16-31). Weight rows are 16B aligned for all
// K used here, so each 8-half chunk is a single b128 load.
__device__ __forceinline__ v16bf load_a_frag(const __bf16* __restrict__ wrow,
                                             int kb, int K) {
  uint4 z4 = {0u, 0u, 0u, 0u};
  uint4 c0 = (kb      < K) ? *(const uint4*)(wrow + kb)      : z4;
  uint4 c1 = (kb + 8  < K) ? *(const uint4*)(wrow + kb + 8)  : z4;
  v8bf lo = __builtin_bit_cast(v8bf, c0);
  v8bf hh = __builtin_bit_cast(v8bf, c1);
  return __builtin_shufflevector(lo, hh, 0, 1, 2, 3, 4, 5, 6, 7,
                                 8, 9, 10, 11, 12, 13, 14, 15);
}

// ---------------- implicit-GEMM conv ----------------
// A = weights [Cout x K] (OIHW flat bf16), B = im2col input, C: M=channel, N=pixel.
// One wave computes a (16*CHT)ch x 32px tile = CHT x 2 WMMA accumulators.
// For 4x4 kernels a 16-element K-chunk is one input channel's patch, so the
// gather offsets/masks are K-invariant and hoisted out of the K-loop.
template <int KH, int KW, int STRIDE, int PAD, bool RELU, bool IN_F32, bool OUT_F32,
          int Cin, int H, int W, int Cout, int OH, int OW, int CHT>
__global__ __launch_bounds__(128)
void conv_wmma_kernel(const void* __restrict__ in_, const __bf16* __restrict__ wbf,
                      const float* __restrict__ bias, void* __restrict__ out_) {
  constexpr int K   = Cin * KH * KW;
  constexpr int HWo = OH * OW;
  constexpr int HWi = H * W;

  const int lane = threadIdx.x & 31;
  const int wave = threadIdx.x >> 5;
  const int mt   = blockIdx.x * 4 + wave;    // 32-pixel tile
  const int n0   = blockIdx.y * (16 * CHT);  // channel tile
  const int m0   = mt * 32;                  // OW % 32 == 0 -> tile shares (b,oy)
  const int b    = m0 / HWo;
  const int rem  = m0 - b * HWo;
  const int oy   = rem / OW;
  const int ox0  = rem - oy * OW;

  const int  nlo = lane & 15;
  const bool hi  = lane >= 16;
  const int  ka0 = hi ? 16 : 0;  // A: lanes 16-31 hold K[+16..31]
  const int  kb0 = hi ? 16 : 0;  // B: lanes 16-31 hold K[+16..31]
  const int  px0 = ox0 + nlo;    // pixel for B subtile 0 (subtile 1: +16)

  const float*  inf32 = (const float*)in_;
  const __bf16* inb16 = (const __bf16*)in_;
  const __bf16  BZ = __builtin_bit_cast(__bf16, (unsigned short)0);
  const size_t  bbase = (size_t)b * Cin * HWi;

  // ---- hoisted im2col addressing for the 4x4 path (K-invariant) ----
  int      boff[2][16];
  unsigned bmask[2] = {0u, 0u};
  if (KH == 4 && KW == 4) {
#pragma unroll
    for (int i = 0; i < 2; ++i) {
      const int px  = px0 + i * 16;
      const int iy0 = oy * STRIDE - PAD;
      const int ix0 = px * STRIDE - PAD;
#pragma unroll
      for (int e = 0; e < 16; ++e) {
        const int  ky = e >> 2, kx = e & 3;
        const int  iy = iy0 + ky, ix = ix0 + kx;
        const bool ok = (iy >= 0) && (iy < H) && (ix >= 0) && (ix < W);
        boff[i][e] = ok ? (iy * W + ix) : 0;
        if (ok) bmask[i] |= (1u << e);
      }
    }
  }

  v8f acc[CHT][2];
#pragma unroll
  for (int j = 0; j < CHT; ++j) { acc[j][0] = (v8f){}; acc[j][1] = (v8f){}; }

  for (int k0 = 0; k0 < K; k0 += 32) {
    const int kbase = k0 + kb0;

    // ---- B fragments: 2 pixel sub-tiles ----
    v16bf bfr[2];
    if (KH == 4 && KW == 4) {
      if (kbase >= K) {                     // K % 16 == 0 -> exact chunk guard
#pragma unroll
        for (int e = 0; e < 16; ++e) { bfr[0][e] = BZ; bfr[1][e] = BZ; }
      } else {
        const int ci = kbase >> 4;          // one chunk == one input channel
#pragma unroll
        for (int i = 0; i < 2; ++i) {
          if (IN_F32) {
            const float* p = inf32 + bbase + (size_t)ci * HWi;
#pragma unroll
            for (int e = 0; e < 16; ++e)
              bfr[i][e] = ((bmask[i] >> e) & 1u) ? f2bf(p[boff[i][e]]) : BZ;
          } else {
            const __bf16* p = inb16 + bbase + (size_t)ci * HWi;
#pragma unroll
            for (int e = 0; e < 16; ++e)
              bfr[i][e] = ((bmask[i] >> e) & 1u) ? p[boff[i][e]] : BZ;
          }
        }
        // warm WGP$ for the next K-step's input channel (global_prefetch_b8)
        if (!IN_F32 && (ci + 2) < Cin)
          __builtin_prefetch(inb16 + bbase + (size_t)(ci + 2) * HWi + boff[0][0], 0, 0);
      }
    } else {
#pragma unroll
      for (int i = 0; i < 2; ++i) {
        const int px = px0 + i * 16;
#pragma unroll
        for (int e = 0; e < 16; ++e) {
          const int k = kbase + e;
          __bf16 v = BZ;
          if (k < K) {
            const int ci = k / (KH * KW);    // constexpr divisor
            const int rr = k - ci * (KH * KW);
            const int ky = rr / KW;
            const int kx = rr - ky * KW;
            const int iy = oy * STRIDE - PAD + ky;
            const int ix = px * STRIDE - PAD + kx;
            if (iy >= 0 && iy < H && ix >= 0 && ix < W) {
              const size_t idx = bbase + (size_t)ci * HWi + (size_t)iy * W + ix;
              v = IN_F32 ? f2bf(inf32[idx]) : inb16[idx];
            }
          }
          bfr[i][e] = v;
        }
      }
    }

    // ---- CHT x 2 WMMAs per K-step, A fragments loaded as b128 pairs ----
#pragma unroll
    for (int j = 0; j < CHT; ++j) {
      const __bf16* wrow = wbf + (size_t)(n0 + j * 16 + nlo) * (size_t)K;
      const v16bf a = load_a_frag(wrow, k0 + ka0, K);
      acc[j][0] = __builtin_amdgcn_wmma_f32_16x16x32_bf16(false, a, false, bfr[0],
                                                          (short)0, acc[j][0], false, false);
      acc[j][1] = __builtin_amdgcn_wmma_f32_16x16x32_bf16(false, a, false, bfr[1],
                                                          (short)0, acc[j][1], false, false);
    }
  }

  // ---- store: C VGPR r -> M = r (lanes 0-15) / r+8 (lanes 16-31), N = pixel ----
#pragma unroll
  for (int j = 0; j < CHT; ++j) {
    const int chb = n0 + j * 16 + (hi ? 8 : 0);
#pragma unroll
    for (int r = 0; r < 8; ++r) {
      const int   ch = chb + r;
      const float bv = bias[ch];
#pragma unroll
      for (int i = 0; i < 2; ++i) {
        float v = acc[j][i][r] + bv;
        if (RELU) v = v > 0.f ? v : 0.f;
        const size_t oidx = (size_t)(b * Cout + ch) * (size_t)HWo +
                            (size_t)oy * OW + (px0 + i * 16);
        if (OUT_F32) ((float*)out_)[oidx] = v;
        else         ((__bf16*)out_)[oidx] = f2bf(v);
      }
    }
  }
}

// ---------------- VQ: LDS-staged codebook, exact f32 argmin ----------------
__global__ __launch_bounds__(256)
void vq_kernel(const float* __restrict__ z, const float* __restrict__ emb,
               float* __restrict__ zq, unsigned* __restrict__ hist,
               float* __restrict__ partial) {
  __shared__ __align__(16) float semb[256 * 64];   // 64 KiB: half the codebook
  __shared__ __align__(16) float red[256];
  const int row = blockIdx.x * 256 + threadIdx.x;  // 65536 rows total

  const float4* zr = (const float4*)(z + (size_t)row * 64);
  float4 zv[16];
#pragma unroll
  for (int i = 0; i < 16; ++i) zv[i] = zr[i];

  // argmin_k ( sum_d e*(e - 2z) )  == argmin of zsq + esq - 2*dot (zsq const)
  float best = 3.4e38f;
  int   bidx = 0;
  for (int half = 0; half < 2; ++half) {
    __syncthreads();  // previous half's readers done
    {
      const float4* src = (const float4*)(emb + half * 256 * 64);
      float4*       dst = (float4*)semb;
      for (int i = threadIdx.x; i < 4096; i += 256) dst[i] = src[i];
    }
    __syncthreads();
    for (int k = 0; k < 256; ++k) {
      const float4* er = (const float4*)(semb + k * 64);
      float s = 0.f;
#pragma unroll
      for (int i = 0; i < 16; ++i) {
        float4 e = er[i], zz = zv[i];
        s = fmaf(e.x, fmaf(-2.f, zz.x, e.x), s);
        s = fmaf(e.y, fmaf(-2.f, zz.y, e.y), s);
        s = fmaf(e.z, fmaf(-2.f, zz.z, e.z), s);
        s = fmaf(e.w, fmaf(-2.f, zz.w, e.w), s);
      }
      if (s < best) { best = s; bidx = half * 256 + k; }  // first-min tie-break
    }
  }
  atomicAdd(&hist[bidx], 1u);

  const float4* eq = (const float4*)(emb + (size_t)bidx * 64);
  float4* out4 = (float4*)(zq + (size_t)row * 64);
  float sse = 0.f;
#pragma unroll
  for (int i = 0; i < 16; ++i) {
    float4 e = eq[i], zz = zv[i], d, o;
    d.x = e.x - zz.x; d.y = e.y - zz.y; d.z = e.z - zz.z; d.w = e.w - zz.w;
    o.x = zz.x + d.x; o.y = zz.y + d.y; o.z = zz.z + d.z; o.w = zz.w + d.w;
    sse += d.x * d.x + d.y * d.y + d.z * d.z + d.w * d.w;
    out4[i] = o;   // z_q = z + stop_grad(q - z)
  }
  red[threadIdx.x] = sse;
  __syncthreads();
  for (int s = 128; s > 0; s >>= 1) {
    if (threadIdx.x < s) red[threadIdx.x] += red[threadIdx.x + s];
    __syncthreads();
  }
  if (threadIdx.x == 0) partial[blockIdx.x] = red[0];
}

// ---------------- finalize: vq_loss + perplexity ----------------
__global__ __launch_bounds__(512)
void finalize_kernel(const float* __restrict__ partial, const unsigned* __restrict__ hist,
                     float* __restrict__ out_scalars) {
  __shared__ float red[512];
  const int t = threadIdx.x;
  red[t] = (t < 256) ? partial[t] : 0.f;
  __syncthreads();
  for (int s = 256; s > 0; s >>= 1) {
    if (t < s) red[t] += red[t + s];
    __syncthreads();
  }
  const float sse = red[0];
  __syncthreads();
  float p = (float)hist[t] * (1.f / 65536.f);
  red[t] = p * logf(p + 1e-10f);
  __syncthreads();
  for (int s = 256; s > 0; s >>= 1) {
    if (t < s) red[t] += red[t + s];
    __syncthreads();
  }
  if (t == 0) {
    // q_latent + BETA*e_latent, both numerically mean((q-z)^2) over 16*64*64*64
    out_scalars[0] = 1.25f * sse / 4194304.f;
    out_scalars[1] = expf(-red[0]);
  }
}

// ---------------- workspace layout (bytes) ----------------
static const size_t OFF_H1   = 0;                       // 33,554,432 bf16 = 64 MiB
static const size_t OFF_H2   = 67108864;                // 16,777,216 bf16 = 32 MiB
static const size_t OFF_Z    = 100663296;               //  4,194,304 f32  = 16 MiB
static const size_t OFF_W1B  = 117440512;               //  6,144 bf16
static const size_t OFF_W2B  = 117452800;               //  524,288 bf16
static const size_t OFF_W3B  = 118501376;               //  147,456 bf16
static const size_t OFF_HIST = 118796288;               //  512 u32
static const size_t OFF_PART = 118798336;               //  256 f32

extern "C" void kernel_launch(void* const* d_in, const int* in_sizes, int n_in,
                              void* d_out, int out_size, void* d_ws, size_t ws_size,
                              hipStream_t stream) {
  (void)in_sizes; (void)n_in; (void)out_size; (void)ws_size;

  const float* x   = (const float*)d_in[0];
  const float* w1  = (const float*)d_in[1];
  const float* b1  = (const float*)d_in[2];
  const float* w2  = (const float*)d_in[3];
  const float* b2  = (const float*)d_in[4];
  const float* w3  = (const float*)d_in[5];
  const float* b3  = (const float*)d_in[6];
  const float* emb = (const float*)d_in[7];
  // d_in[8..13] (decoder weights) are dead code: reference returns only
  // (z_q, vq_loss, perplexity) and the decoder output is unused.

  float* out = (float*)d_out;   // [0..4194303]=z_q, [4194304]=loss, [4194305]=perp
  char*  ws  = (char*)d_ws;

  __bf16*   h1   = (__bf16*)(ws + OFF_H1);
  __bf16*   h2   = (__bf16*)(ws + OFF_H2);
  float*    zbuf = (float*)(ws + OFF_Z);
  __bf16*   w1b  = (__bf16*)(ws + OFF_W1B);
  __bf16*   w2b  = (__bf16*)(ws + OFF_W2B);
  __bf16*   w3b  = (__bf16*)(ws + OFF_W3B);
  unsigned* hist = (unsigned*)(ws + OFF_HIST);
  float*    part = (float*)(ws + OFF_PART);

  // zero the accumulators every call (deterministic)
  hipMemsetAsync(ws + OFF_HIST, 0, 512 * 4 + 256 * 4, stream);

  // weights -> bf16 (OIHW flat == [Cout][K] GEMM-A layout)
  cvt_bf16_kernel<<<(6144   + 255) / 256, 256, 0, stream>>>(w1, w1b, 6144);
  cvt_bf16_kernel<<<(524288 + 255) / 256, 256, 0, stream>>>(w2, w2b, 524288);
  cvt_bf16_kernel<<<(147456 + 255) / 256, 256, 0, stream>>>(w3, w3b, 147456);

  // encoder: conv1 4x4/s2 relu, conv2 4x4/s2 relu, conv3 3x3/s1
  // grid.x = (B*OH*OW)/32/4, grid.y = Cout/(16*CHT)
  conv_wmma_kernel<4, 4, 2, 1, true, true, false, 3, 256, 256, 128, 128, 128, 8>
      <<<dim3(2048, 1), 128, 0, stream>>>(x, w1b, b1, h1);
  conv_wmma_kernel<4, 4, 2, 1, true, false, false, 128, 128, 128, 256, 64, 64, 8>
      <<<dim3(512, 2), 128, 0, stream>>>(h1, w2b, b2, h2);
  conv_wmma_kernel<3, 3, 1, 1, false, false, true, 256, 64, 64, 64, 64, 64, 4>
      <<<dim3(512, 1), 128, 0, stream>>>(h2, w3b, b3, zbuf);

  // vector quantizer + straight-through output + stats
  vq_kernel<<<256, 256, 0, stream>>>(zbuf, emb, out, hist, part);
  finalize_kernel<<<1, 512, 0, stream>>>(part, hist, out + 4194304);
}
